// GeometricProductBase_40965398069480
// MI455X (gfx1250) — matrix operand order repaired and checked by reading, not compiled, not run on hardware
//
#include <hip/hip_runtime.h>
#include <hip/hip_bf16.h>
#include <math.h>

// ---------------------------------------------------------------------------
// Types for CDNA5 WMMA
// ---------------------------------------------------------------------------
typedef __attribute__((ext_vector_type(8)))  float   v8f;
typedef __attribute__((ext_vector_type(16))) __bf16  v16bf;
typedef __attribute__((ext_vector_type(8)))  __bf16  v8bf;

#define FDIM 512
#define NBL  8

// bf16 conversion helpers (round-to-nearest-even), bit-level.
__device__ inline __bf16 f2bf(float v) {
    union { float f; unsigned u; } a; a.f = v;
    unsigned r = (a.u + 0x7fffu + ((a.u >> 16) & 1u)) >> 16;
    union { unsigned short s; __bf16 b; } o; o.s = (unsigned short)r;
    return o.b;
}
__device__ inline float bf2f(__bf16 b) {
    union { unsigned short s; __bf16 b; } i; i.b = b;
    union { unsigned u; float f; } o; o.u = ((unsigned)i.s) << 16;
    return o.f;
}

// Build a v16bf fragment from two 16-byte LDS chunks (ds_load_b128 each).
__device__ inline v16bf ld_frag(const __bf16* p0, const __bf16* p1) {
    v8bf a = *(const v8bf*)p0;
    v8bf b = *(const v8bf*)p1;
    return __builtin_shufflevector(a, b, 0,1,2,3,4,5,6,7,8,9,10,11,12,13,14,15);
}

// gfx1250 async global->LDS copy (ASYNCcnt-tracked DMA, no VGPR round trip).
// Generic LDS addresses carry the LDS byte offset in their low 32 bits.
__device__ inline unsigned lds_off(const void* p) {
    return (unsigned)(uintptr_t)p;
}
__device__ inline void async_b128(unsigned lds, const __bf16* g) {
    asm volatile("global_load_async_to_lds_b128 %0, %1, off"
                 :: "v"(lds), "v"((unsigned long long)(uintptr_t)g)
                 : "memory");
}
__device__ inline void wait_async0() {
    asm volatile("s_wait_asynccnt 0x0" ::: "memory");
}

// ---------------------------------------------------------------------------
// Pre-pass 1: split + transpose x  [B,F,8] f32  ->  xh/xl [8][B][F] bf16
// ---------------------------------------------------------------------------
__global__ __launch_bounds__(256)
void split_x_kernel(const float* __restrict__ x,
                    __bf16* __restrict__ xh, __bf16* __restrict__ xl,
                    int Bsz) {
    const int fq = FDIM / 4;
    const int t  = blockIdx.x * 256 + threadIdx.x;
    const int total = NBL * Bsz * fq;
    if (t >= total) return;
    const int f4 = (t % fq) * 4;
    const int b  = (t / fq) % Bsz;
    const int m  = t / (fq * Bsz);
    const float* src = x + ((size_t)b * FDIM + f4) * NBL + m;
    const size_t d = ((size_t)m * Bsz + b) * FDIM + f4;
    #pragma unroll
    for (int j = 0; j < 4; ++j) {
        const float v = src[j * NBL];
        const __bf16 h = f2bf(v);
        xh[d + j] = h;
        xl[d + j] = f2bf(v - bf2f(h));
    }
}

// ---------------------------------------------------------------------------
// Pre-pass 2: split both weight tensors [2,F,F] f32 -> bf16 hi/lo
// ---------------------------------------------------------------------------
__global__ __launch_bounds__(256)
void split_w_kernel(const float* __restrict__ wr, const float* __restrict__ wl,
                    __bf16* __restrict__ wrh, __bf16* __restrict__ wrl,
                    __bf16* __restrict__ wlh, __bf16* __restrict__ wll) {
    const int t  = blockIdx.x * 256 + threadIdx.x;   // < 2*F*F/4
    const size_t i4 = (size_t)t * 4;
    const float4 a = *(const float4*)(wr + i4);
    const float4 b = *(const float4*)(wl + i4);
    const float av[4] = { a.x, a.y, a.z, a.w };
    const float bv[4] = { b.x, b.y, b.z, b.w };
    #pragma unroll
    for (int j = 0; j < 4; ++j) {
        __bf16 h = f2bf(av[j]);
        wrh[i4 + j] = h;
        wrl[i4 + j] = f2bf(av[j] - bf2f(h));
        h = f2bf(bv[j]);
        wlh[i4 + j] = h;
        wll[i4 + j] = f2bf(bv[j] - bf2f(h));
    }
}

// ---------------------------------------------------------------------------
// Parity-linear GEMM:  y[b,o,m] = sum_f w[p_m][o,f] * x[b,f,m]  (+ bias[o,m])
// A/B pre-split to bf16 hi/lo; C ~= Ah*Bh + Ah*Bl + Al*Bh (near-fp32).
// Double-buffered LDS fed by async global->LDS b128 DMA.
// ---------------------------------------------------------------------------
#define TM 64
#define TN 128
#define TK 32
#define LDA 40   // padded row stride (bf16 elems): 80B rows, 16B aligned
#define LDB 40
#define NK  (FDIM / TK)

template <bool HAS_BIAS>
__global__ __launch_bounds__(256)
void plinear_wmma(const __bf16* __restrict__ xh,   // [8][B][F]
                  const __bf16* __restrict__ xl,   // [8][B][F]
                  const __bf16* __restrict__ wh,   // [2][F][F]
                  const __bf16* __restrict__ wl,   // [2][F][F]
                  const float*  __restrict__ bias, // [F,8] or nullptr
                  float* __restrict__ y,           // [B, F, 8]
                  int Bsz)
{
    __shared__ alignas(16) __bf16 sAh[2][TM * LDA];
    __shared__ alignas(16) __bf16 sAl[2][TM * LDA];
    __shared__ alignas(16) __bf16 sBh[2][TN * LDB];
    __shared__ alignas(16) __bf16 sBl[2][TN * LDB];

    const int m   = blockIdx.z;
    const int par = __popc(m) & 1;
    const __bf16* xAh = xh + (size_t)m * Bsz * FDIM;
    const __bf16* xAl = xl + (size_t)m * Bsz * FDIM;
    const __bf16* wBh = wh + (size_t)par * FDIM * FDIM;
    const __bf16* wBl = wl + (size_t)par * FDIM * FDIM;

    const int bm0  = blockIdx.x * TM;
    const int on0  = blockIdx.y * TN;

    const int tid  = threadIdx.x;
    const int lane = tid & 31;
    const int wave = tid >> 5;
    const int wm   = wave & 1;   // 2 wave-rows  (32 batch rows each)
    const int wn   = wave >> 1;  // 4 wave-cols  (32 out cols each)
    const int half = lane >> 4;
    const int l16  = lane & 15;

    // per-thread staging assignment (pure DMA, no VALU in the hot loop)
    const int ar = tid >> 2;            // A row 0..63
    const int ac = (tid & 3) * 8;       // A k-chunk 0,8,16,24
    const int br = tid >> 1;            // B row 0..127
    const int bc = (tid & 1) * 16;      // B k-chunk 0 or 16

    const __bf16* gAh = xAh + (size_t)(bm0 + ar) * FDIM + ac;
    const __bf16* gAl = xAl + (size_t)(bm0 + ar) * FDIM + ac;
    const __bf16* gBh = wBh + (size_t)(on0 + br) * FDIM + bc;
    const __bf16* gBl = wBl + (size_t)(on0 + br) * FDIM + bc;
    const int liA = ar * LDA + ac;
    const int liB = br * LDB + bc;

    auto stage = [&](int buf, int k0) {
        async_b128(lds_off(&sAh[buf][liA]), gAh + k0);
        async_b128(lds_off(&sAl[buf][liA]), gAl + k0);
        async_b128(lds_off(&sBh[buf][liB]),     gBh + k0);
        async_b128(lds_off(&sBh[buf][liB + 8]), gBh + k0 + 8);
        async_b128(lds_off(&sBl[buf][liB]),     gBl + k0);
        async_b128(lds_off(&sBl[buf][liB + 8]), gBl + k0 + 8);
    };

    v8f acc[2][2] = {};  // 2x2 tiles of 16x16 fp32

    stage(0, 0);
    for (int kk = 0; kk < NK; ++kk) {
        const int cur = kk & 1;
        wait_async0();
        __syncthreads();                       // buf[cur] ready for all waves
        if (kk + 1 < NK) stage(cur ^ 1, (kk + 1) * TK);  // overlap DMA w/ WMMA

        // ---- gather fragments per documented 16-bit WMMA layouts ----------
        v16bf ah[2], al[2], bh[2], bl[2];
        #pragma unroll
        for (int mt = 0; mt < 2; ++mt) {
            const int a0 = (wm * 32 + mt * 16 + l16) * LDA;
            ah[mt] = ld_frag(&sAh[cur][a0 + half * 8], &sAh[cur][a0 + 16 + half * 8]);
            al[mt] = ld_frag(&sAl[cur][a0 + half * 8], &sAl[cur][a0 + 16 + half * 8]);
        }
        #pragma unroll
        for (int nt = 0; nt < 2; ++nt) {
            const int b0 = (wn * 32 + nt * 16 + l16) * LDB;
            bh[nt] = ld_frag(&sBh[cur][b0 + half * 8], &sBh[cur][b0 + 16 + half * 8]);
            bl[nt] = ld_frag(&sBl[cur][b0 + half * 8], &sBl[cur][b0 + 16 + half * 8]);
        }

        #pragma unroll
        for (int mt = 0; mt < 2; ++mt) {
            #pragma unroll
            for (int nt = 0; nt < 2; ++nt) {
                acc[mt][nt] = __builtin_amdgcn_wmma_f32_16x16x32_bf16(
                    false, ah[mt], false, bh[nt], (short)0, acc[mt][nt], false, false);
                acc[mt][nt] = __builtin_amdgcn_wmma_f32_16x16x32_bf16(
                    false, ah[mt], false, bl[nt], (short)0, acc[mt][nt], false, false);
                acc[mt][nt] = __builtin_amdgcn_wmma_f32_16x16x32_bf16(
                    false, al[mt], false, bh[nt], (short)0, acc[mt][nt], false, false);
            }
        }
    }

    // ---- epilogue: C layout VGPR r -> M=r (lanes 0-15) / M=r+8 (16-31) ----
    #pragma unroll
    for (int mt = 0; mt < 2; ++mt) {
        #pragma unroll
        for (int nt = 0; nt < 2; ++nt) {
            const int o  = on0 + wn * 32 + nt * 16 + l16;
            const float bb = HAS_BIAS ? bias[o * NBL + m] : 0.0f;
            #pragma unroll
            for (int r = 0; r < 8; ++r) {
                const int brow = bm0 + wm * 32 + mt * 16 + r + half * 8;
                y[((size_t)brow * FDIM + o) * NBL + m] = acc[mt][nt][r] + bb;
            }
        }
    }
}

// ---------------------------------------------------------------------------
// Fused: pnorm(right-GEMM output) -> x_right, sparse Cayley geo, combine.
// out[b,n,j] = (left[b,n,j] + geo[b,n,j]) / sqrt(2)
// ---------------------------------------------------------------------------
__global__ __launch_bounds__(256)
void combine_kernel(const float* __restrict__ x,       // [B,F,8]
                    const float* __restrict__ yr,      // [B,F,8] raw right GEMM
                    const float* __restrict__ w_prod,  // [F,4]
                    const float* __restrict__ a_norm,  // [F,2]
                    float* __restrict__ out,           // [B,F,8] holds left, rewritten
                    int total)
{
    const int idx = blockIdx.x * 256 + threadIdx.x;
    if (idx >= total) return;
    const int n = idx % FDIM;

    const float* yp = yr  + (size_t)idx * NBL;
    const float* xp = x   + (size_t)idx * NBL;
    float*       op = out + (size_t)idx * NBL;

    float yv[8], xv[8], lf[8];
    {
        const float4 a0 = *(const float4*)(yp);
        const float4 a1 = *(const float4*)(yp + 4);
        yv[0]=a0.x; yv[1]=a0.y; yv[2]=a0.z; yv[3]=a0.w;
        yv[4]=a1.x; yv[5]=a1.y; yv[6]=a1.z; yv[7]=a1.w;
        const float4 b0 = *(const float4*)(xp);
        const float4 b1 = *(const float4*)(xp + 4);
        xv[0]=b0.x; xv[1]=b0.y; xv[2]=b0.z; xv[3]=b0.w;
        xv[4]=b1.x; xv[5]=b1.y; xv[6]=b1.z; xv[7]=b1.w;
        const float4 c0 = *(const float4*)(op);
        const float4 c1 = *(const float4*)(op + 4);
        lf[0]=c0.x; lf[1]=c0.y; lf[2]=c0.z; lf[3]=c0.w;
        lf[4]=c1.x; lf[5]=c1.y; lf[6]=c1.z; lf[7]=c1.w;
    }

    const int PAR[8] = {0,1,1,0,1,0,0,1};
    const int SGN[8][8] = {
        { 1, 1, 1, 1, 1, 1, 1, 1},
        { 1, 1, 1, 1, 1, 1, 1, 1},
        { 1,-1, 1,-1, 1,-1, 1,-1},
        { 1,-1, 1,-1, 1,-1, 1,-1},
        { 1,-1,-1, 1, 1,-1,-1, 1},
        { 1,-1,-1, 1, 1,-1,-1, 1},
        { 1, 1,-1,-1, 1, 1,-1,-1},
        { 1, 1,-1,-1, 1, 1,-1,-1},
    };

    float ssq[2] = {1e-12f, 1e-12f};
    #pragma unroll
    for (int mm = 0; mm < 8; ++mm) ssq[PAR[mm]] += yv[mm] * yv[mm];
    float s[2];
    #pragma unroll
    for (int p = 0; p < 2; ++p) {
        const float a   = a_norm[n * 2 + p];
        const float sig = 1.0f / (1.0f + __expf(-a));
        s[p] = sig * (sqrtf(ssq[p]) - 1.0f) + 1.0f;
    }
    float xr[8];
    #pragma unroll
    for (int mm = 0; mm < 8; ++mm) xr[mm] = yv[mm] / s[PAR[mm]];

    const float wp4[4] = { w_prod[n*4+0], w_prod[n*4+1], w_prod[n*4+2], w_prod[n*4+3] };

    const float inv_sqrt2 = 0.70710678118654752440f;
    float res[8];
    #pragma unroll
    for (int j = 0; j < 8; ++j) {
        float g = 0.0f;
        #pragma unroll
        for (int i = 0; i < 8; ++i) {
            const int k = i ^ j;
            g += (float)SGN[i][k] * wp4[2 * PAR[i] + PAR[k]] * xv[i] * xr[k];
        }
        res[j] = (lf[j] + g) * inv_sqrt2;
    }

    float4 r0; r0.x=res[0]; r0.y=res[1]; r0.z=res[2]; r0.w=res[3];
    float4 r1; r1.x=res[4]; r1.y=res[5]; r1.z=res[6]; r1.w=res[7];
    *(float4*)(op)     = r0;
    *(float4*)(op + 4) = r1;
}

// ---------------------------------------------------------------------------
extern "C" void kernel_launch(void* const* d_in, const int* in_sizes, int n_in,
                              void* d_out, int out_size, void* d_ws, size_t ws_size,
                              hipStream_t stream) {
    const float* x       = (const float*)d_in[0];  // [B, F, 8]
    const float* w_prod  = (const float*)d_in[1];  // [F, 4]
    const float* w_right = (const float*)d_in[2];  // [2, F, F]
    const float* w_left  = (const float*)d_in[3];  // [2, F, F]
    const float* b_left  = (const float*)d_in[4];  // [F, 8]
    const float* a_norm  = (const float*)d_in[5];  // [F, 2]

    const int Bsz = in_sizes[0] / (FDIM * NBL);    // 2048

    // workspace carve-up (bytes)
    char* ws = (char*)d_ws;
    const size_t yBytes  = (size_t)Bsz * FDIM * NBL * sizeof(float);   // 33.5 MB
    const size_t xBytes  = (size_t)NBL * Bsz * FDIM * sizeof(__bf16);  // 16.8 MB
    const size_t wBytes  = (size_t)2 * FDIM * FDIM * sizeof(__bf16);   // 1 MB
    float*  yws = (float*)(ws);
    __bf16* xh  = (__bf16*)(ws + yBytes);
    __bf16* xl  = (__bf16*)(ws + yBytes + xBytes);
    __bf16* wrh = (__bf16*)(ws + yBytes + 2 * xBytes);
    __bf16* wrl = (__bf16*)(ws + yBytes + 2 * xBytes + wBytes);
    __bf16* wlh = (__bf16*)(ws + yBytes + 2 * xBytes + 2 * wBytes);
    __bf16* wll = (__bf16*)(ws + yBytes + 2 * xBytes + 3 * wBytes);

    float* outp = (float*)d_out;

    // pre-split passes
    {
        const int tot = NBL * Bsz * (FDIM / 4);
        split_x_kernel<<<(tot + 255) / 256, 256, 0, stream>>>(x, xh, xl, Bsz);
    }
    {
        const int tot = 2 * FDIM * FDIM / 4;
        split_w_kernel<<<tot / 256, 256, 0, stream>>>(w_right, w_left,
                                                      wrh, wrl, wlh, wll);
    }

    dim3 blk(256);
    dim3 grd(Bsz / TM, FDIM / TN, NBL);
    // right parity-linear -> workspace (pnorm deferred to combine)
    plinear_wmma<false><<<grd, blk, 0, stream>>>(xh, xl, wrh, wrl, nullptr, yws, Bsz);
    // left parity-linear + bias -> d_out
    plinear_wmma<true><<<grd, blk, 0, stream>>>(xh, xl, wlh, wll, b_left, outp, Bsz);
    // fused pnorm + geometric product + combine
    const int total = Bsz * FDIM;
    combine_kernel<<<(total + 255) / 256, 256, 0, stream>>>(x, yws, w_prod, a_norm,
                                                            outp, total);
}
